// KernelAttention_5136780886521
// MI455X (gfx1250) — matrix operand order repaired and checked by reading, not compiled, not run on hardware
//
#include <hip/hip_runtime.h>
#include <hip/hip_bf16.h>

// MHA block for MI455X (gfx1250): fp32 in/out, bf16 WMMA compute, f32 accum.
// GEMMs stage weight tiles via the Tensor Data Mover (tensor_load_to_lds,
// TENSORcnt); attention stages K/V via global_load_async_to_lds_b128
// (ASYNCcnt). Both double-buffered in LDS. Needs 48 MiB workspace.

typedef __attribute__((ext_vector_type(16))) __bf16   v16bf;
typedef __attribute__((ext_vector_type(8)))  __bf16   v8bf;
typedef __attribute__((ext_vector_type(8)))  float    v8f;
typedef __attribute__((ext_vector_type(4)))  unsigned v4u;
typedef __attribute__((ext_vector_type(8)))  unsigned v8u;

#define DIM   1024
#define HEADS 16
#define DH    64
#define SEQ   2048
#define BATCH 2
#define MTOT  (BATCH * SEQ)   // 4096

__device__ __forceinline__ unsigned short f2bf(float f) {
    unsigned int u = __float_as_uint(f);
    u += 0x7FFFu + ((u >> 16) & 1u);   // round-to-nearest-even
    return (unsigned short)(u >> 16);
}

union FragBF { v16bf v; v8bf h[2]; };

// Low 32 bits of the flat address of a __shared__ object == LDS byte offset.
__device__ __forceinline__ unsigned lds_off(const void* p) {
    return (unsigned)(uintptr_t)p;
}

// Async DMA: 16B global -> LDS, tracked by ASYNCcnt (no VGPR round trip).
__device__ __forceinline__ void async_b128(unsigned lds_byte_off, const void* g) {
    asm volatile("global_load_async_to_lds_b128 %0, %1, off"
                 :: "v"(lds_byte_off), "v"(g) : "memory");
}
__device__ __forceinline__ void wait_async(int n) {
    if (n == 0) asm volatile("s_wait_asynccnt 0" ::: "memory");
    else        asm volatile("s_wait_asynccnt 4" ::: "memory");
}

// Tensor Data Mover: 2D tile -> LDS. D# group0 packs {count=1, lds_addr,
// global_addr[56:0], type=2}; group1 carries dims/strides. <=2D tensors use
// the two-SGPR-group form (VADDR2/3 = NULL). Tracked by TENSORcnt.
__device__ __forceinline__ void tdm_load_2d(unsigned lds_addr, unsigned long long ga, v8u g1) {
    v4u g0;
    g0.x = 1u;                                                  // count=1, user mode
    g0.y = lds_addr;                                            // bits 63:32
    g0.z = (unsigned)ga;                                        // bits 95:64
    g0.w = (unsigned)((ga >> 32) & 0x1FFFFFFu) | 0x80000000u;   // addr[56:32] | type=2
    asm volatile("tensor_load_to_lds %0, %1" :: "s"(g0), "s"(g1) : "memory");
}
// group1 for a [nrows x 1024] bf16 tensor, 64row x 32col tile, row stride 1024
__device__ __forceinline__ v8u tdm_g1_w64x32(unsigned nrows) {
    v8u g1 = { 0x00010000u,               // data_size = 2B
               (1024u & 0xFFFFu) << 16,   // tensor_dim0 = 1024 (lo16)
               (nrows & 0xFFFFu) << 16,   // tensor_dim0 hi=0 | tensor_dim1 lo16
               32u << 16,                 // tensor_dim1 hi=0 | tile_dim0 = 32
               64u,                       // tile_dim1 = 64, tile_dim2 = 0
               1024u,                     // tensor_dim0_stride lo32
               0u, 0u };
    return g1;
}
// SIMM16 must be an integer constant expression -> template parameter.
template <int N>
__device__ __forceinline__ void wait_tensor() {
    __builtin_amdgcn_s_wait_tensorcnt(N);
}

__global__ void cvt_f32_bf16(const float* __restrict__ in,
                             unsigned short* __restrict__ out, int n) {
    int i = blockIdx.x * blockDim.x + threadIdx.x;
    if (i < n) out[i] = f2bf(in[i]);
}

// ---------------------------------------------------------------------------
// Staged GEMM: block = 8 waves = 128 rows x 64 cols. B tile (64n x 32k bf16,
// 4 KB) is TDM-DMA'd into LDS once per k-step by wave 0 and shared by all 8
// waves. A fragments stream from global (per-wave-private rows). Fragment
// addressing follows the CDNA5 16-bit WMMA VGPR layouts.
// ---------------------------------------------------------------------------
__device__ __forceinline__ void gemm_tile_staged(
    const unsigned short* __restrict__ A,
    const unsigned short* __restrict__ Bm,
    unsigned short (*Bsm)[64 * 32],       // [2][2048] double buffer
    int m0, int n0, int nrows, int tid, v8f acc[4]) {
    const int lane = tid & 31, wave = tid >> 5;
    const int l16  = lane & 15;
    const int half = lane >> 4;
    const unsigned short* arow = A + (size_t)(m0 + l16) * DIM;
    const v8u g1 = tdm_g1_w64x32((unsigned)nrows);
    const unsigned ldsb[2] = { lds_off(&Bsm[0][0]), lds_off(&Bsm[1][0]) };
    const unsigned long long gbase =
        (unsigned long long)(uintptr_t)(Bm + (size_t)n0 * DIM);

    if (wave == 0) tdm_load_2d(ldsb[0], gbase, g1);        // prefetch k0 = 0
    for (int k0 = 0; k0 < DIM; k0 += 32) {
        const int cur = (k0 >> 5) & 1;
        if (wave == 0) {
            if (k0 + 32 < DIM) {
                tdm_load_2d(ldsb[cur ^ 1], gbase + (size_t)(k0 + 32) * 2, g1);
                wait_tensor<1>();                          // current tile landed
            } else {
                wait_tensor<0>();
            }
        }
        __syncthreads();
        FragBF a;
        a.h[0] = *(const v8bf*)(arow + k0 + half * 8);
        a.h[1] = *(const v8bf*)(arow + k0 + 16 + half * 8);
        const unsigned short* bb = Bsm[cur];
        FragBF b[4];
        #pragma unroll
        for (int j = 0; j < 4; ++j) {
            const unsigned short* brow = bb + (j * 16 + l16) * 32 + half * 8;
            b[j].h[0] = *(const v8bf*)(brow);
            b[j].h[1] = *(const v8bf*)(brow + 16);
        }
        #pragma unroll
        for (int j = 0; j < 4; ++j)
            acc[j] = __builtin_amdgcn_wmma_f32_16x16x32_bf16(
                false, a.v, false, b[j].v, (short)0, acc[j], false, false);
        __syncthreads();                                   // done reading buf[cur]
    }
}

// qkv = x @ w_qkv^T, scattered into Q,K:[B,H,L,Dh] and V transposed:[B,H,Dh,L]
__global__ void __launch_bounds__(256) gemm_qkv(
    const unsigned short* __restrict__ xbf,
    const unsigned short* __restrict__ wbf,
    unsigned short* __restrict__ qb,
    unsigned short* __restrict__ kb,
    unsigned short* __restrict__ vb) {
    __shared__ unsigned short Bsm[2][64 * 32];
    const int tid = threadIdx.x;
    const int lane = tid & 31, wave = tid >> 5;
    const int m0 = blockIdx.x * 128 + wave * 16;
    const int n0 = blockIdx.y * 64;
    v8f acc[4] = {};
    gemm_tile_staged(xbf, wbf, Bsm, m0, n0, 3 * DIM, tid, acc);
    const int l16 = lane & 15, half = lane >> 4;
    #pragma unroll
    for (int j = 0; j < 4; ++j) {
        const int col   = n0 + j * 16 + l16;      // [0, 3072)
        const int which = col >> 10;              // 0=q 1=k 2=v
        const int c = col & 1023, h = c >> 6, d = c & 63;
        #pragma unroll
        for (int i = 0; i < 8; ++i) {
            const int row = m0 + i + half * 8;    // [0, 4096)
            const int b = row >> 11, l = row & 2047;
            const unsigned short bf = f2bf(acc[j][i]);
            const size_t qk = ((size_t)((b * HEADS + h) * SEQ + l)) * DH + d;
            if (which == 0)      qb[qk] = bf;
            else if (which == 1) kb[qk] = bf;
            else vb[((size_t)((b * HEADS + h) * DH + d)) * SEQ + l] = bf;
        }
    }
}

// ---------------------------------------------------------------------------
// Flash attention. Block = 4 waves covering 4 adjacent 16-query tiles of ONE
// (b,h); K/V 32-key tiles (4 KB + 4 KB) async-DMA'd into LDS per step and
// shared by all 4 waves. Per wave:
//   S^T = K_tile @ Q^T  (softmax key-reduction = in-lane + one shfl_xor(16))
//   O^T += V^T @ P      (P bounced through per-wave LDS as bf16 B-fragments)
// ---------------------------------------------------------------------------
__global__ void __launch_bounds__(128) attn_kernel(
    const unsigned short* __restrict__ qbuf,
    const unsigned short* __restrict__ kbuf,
    const unsigned short* __restrict__ vbuf,
    unsigned short* __restrict__ obuf) {
    __shared__ unsigned short Ksm[2][32 * 64];   // [key][feat]
    __shared__ unsigned short Vsm[2][64 * 32];   // [d][key]  (V already ^T)
    __shared__ unsigned short Pb[4][16 * 32];    // per-wave P tile
    const int tid = threadIdx.x;
    const int lane = tid & 31, wave = tid >> 5;
    const int t  = blockIdx.x * 4 + wave;
    const int b  = t >> 11;                      // 4 waves share (b,h):
    const int h  = (t >> 7) & 15;                // 128 q-tiles/head, 4 | 128
    const int q0 = (t & 127) << 4;
    const int l16 = lane & 15, half = lane >> 4;

    const unsigned short* qb  = qbuf + ((size_t)((b * HEADS + h) * SEQ + q0)) * DH;
    const unsigned short* kbp = kbuf + (size_t)(b * HEADS + h) * SEQ * DH;
    const unsigned short* vbp = vbuf + (size_t)(b * HEADS + h) * DH * SEQ;

    // producers: thread t owns elems [t*16, t*16+16) of each 2048-elem tile
    const unsigned short* ksrc = kbp + (size_t)(tid >> 2) * DH + (tid & 3) * 16;
    const unsigned short* vsrc = vbp + (size_t)(tid >> 1) * SEQ + (tid & 1) * 16;
    const unsigned kls[2] = { lds_off(&Ksm[0][tid * 16]), lds_off(&Ksm[1][tid * 16]) };
    const unsigned vls[2] = { lds_off(&Vsm[0][tid * 16]), lds_off(&Vsm[1][tid * 16]) };

    // Q as B-fragments (N=query, K=feature), loaded once, reused 64x
    FragBF bq[2];
    const unsigned short* qrow = qb + l16 * DH;
    bq[0].h[0] = *(const v8bf*)(qrow + half * 8);
    bq[0].h[1] = *(const v8bf*)(qrow + 16 + half * 8);
    bq[1].h[0] = *(const v8bf*)(qrow + 32 + half * 8);
    bq[1].h[1] = *(const v8bf*)(qrow + 48 + half * 8);

    v8f o[4] = {};
    float m_i = -3.4e38f, l_i = 0.f;
    const float scale = 0.03125f;    // C^-0.5 = 1/32 (full dim, as reference)
    unsigned short* Pw = Pb[wave];
    const int koff = half * 8;

    // prefetch first K/V tiles (4 x b128 per thread)
    async_b128(kls[0],      ksrc);
    async_b128(kls[0] + 16, ksrc + 8);
    async_b128(vls[0],      vsrc);
    async_b128(vls[0] + 16, vsrc + 8);

    for (int k0 = 0; k0 < SEQ; k0 += 32) {
        const int cur = (k0 >> 5) & 1;
        if (k0 + 32 < SEQ) {
            const int nxt = cur ^ 1;
            const unsigned short* kn = ksrc + (size_t)(k0 + 32) * DH;
            const unsigned short* vn = vsrc + k0 + 32;
            async_b128(kls[nxt],      kn);
            async_b128(kls[nxt] + 16, kn + 8);
            async_b128(vls[nxt],      vn);
            async_b128(vls[nxt] + 16, vn + 8);
            wait_async(4);                       // current 4 landed
        } else {
            wait_async(0);
        }
        __syncthreads();

        // S^T tiles (keys kt*16.., queries) from staged K
        v8f s[2];
        #pragma unroll
        for (int kt = 0; kt < 2; ++kt) {
            const unsigned short* krow = Ksm[cur] + (kt * 16 + l16) * 64;
            FragBF a0, a1;
            a0.h[0] = *(const v8bf*)(krow + half * 8);
            a0.h[1] = *(const v8bf*)(krow + 16 + half * 8);
            a1.h[0] = *(const v8bf*)(krow + 32 + half * 8);
            a1.h[1] = *(const v8bf*)(krow + 48 + half * 8);
            v8f z = {};
            z = __builtin_amdgcn_wmma_f32_16x16x32_bf16(
                false, a0.v, false, bq[0].v, (short)0, z, false, false);
            z = __builtin_amdgcn_wmma_f32_16x16x32_bf16(
                false, a1.v, false, bq[1].v, (short)0, z, false, false);
            s[kt] = z;
        }
        // online softmax (per-lane query; combine lane halves with one shuffle)
        float mloc = -3.4e38f;
        #pragma unroll
        for (int kt = 0; kt < 2; ++kt)
            #pragma unroll
            for (int i = 0; i < 8; ++i) {
                float v = s[kt][i] * scale;
                s[kt][i] = v;
                mloc = fmaxf(mloc, v);
            }
        mloc = fmaxf(mloc, __shfl_xor(mloc, 16, 32));
        const float mnew  = fmaxf(m_i, mloc);
        const float alpha = __expf(m_i - mnew);
        float psum = 0.f;
        #pragma unroll
        for (int kt = 0; kt < 2; ++kt)
            #pragma unroll
            for (int i = 0; i < 8; ++i) {
                float p = __expf(s[kt][i] - mnew);
                psum += p;
                Pw[l16 * 32 + kt * 16 + koff + i] = f2bf(p);
            }
        psum += __shfl_xor(psum, 16, 32);
        l_i = l_i * alpha + psum;
        m_i = mnew;
        #pragma unroll
        for (int tt = 0; tt < 4; ++tt)
            #pragma unroll
            for (int i = 0; i < 8; ++i) o[tt][i] *= alpha;

        // LDS is in-order per wave: drain DS stores before cross-lane reads
        asm volatile("s_wait_dscnt 0" ::: "memory");

        FragBF bp;   // P as B-fragment (N=query, K=32 keys)
    bp.h[0] = *(const v8bf*)(Pw + l16 * 32 + half * 8);
        bp.h[1] = *(const v8bf*)(Pw + l16 * 32 + 16 + half * 8);

        #pragma unroll
        for (int tt = 0; tt < 4; ++tt) {
            const unsigned short* vrow = Vsm[cur] + (tt * 16 + l16) * 32;
            FragBF a;
            a.h[0] = *(const v8bf*)(vrow + half * 8);
            a.h[1] = *(const v8bf*)(vrow + 16 + half * 8);
            o[tt] = __builtin_amdgcn_wmma_f32_16x16x32_bf16(
                false, a.v, false, bp.v, (short)0, o[tt], false, false);
        }
        __syncthreads();                         // done reading buf[cur]
    }

    const float rl = 1.f / l_i;
    #pragma unroll
    for (int tt = 0; tt < 4; ++tt)
        #pragma unroll
        for (int i = 0; i < 8; ++i) {
            const int d = tt * 16 + i + half * 8;
            const int q = q0 + l16;
            obuf[((size_t)(b * SEQ + q)) * DIM + h * DH + d] = f2bf(o[tt][i] * rl);
        }
}

// out = attn @ w_out^T + b_out (fp32 store)
__global__ void __launch_bounds__(256) gemm_out(
    const unsigned short* __restrict__ abf,
    const unsigned short* __restrict__ wbf,
    const float* __restrict__ bias,
    float* __restrict__ out) {
    __shared__ unsigned short Bsm[2][64 * 32];
    const int tid = threadIdx.x;
    const int lane = tid & 31, wave = tid >> 5;
    const int m0 = blockIdx.x * 128 + wave * 16;
    const int n0 = blockIdx.y * 64;
    v8f acc[4] = {};
    gemm_tile_staged(abf, wbf, Bsm, m0, n0, DIM, tid, acc);
    const int l16 = lane & 15, half = lane >> 4;
    #pragma unroll
    for (int j = 0; j < 4; ++j) {
        const int col = n0 + j * 16 + l16;
        const float bv = bias[col];
        #pragma unroll
        for (int i = 0; i < 8; ++i) {
            const int row = m0 + i + half * 8;
            out[(size_t)row * DIM + col] = acc[j][i] + bv;
        }
    }
}

extern "C" void kernel_launch(void* const* d_in, const int* in_sizes, int n_in,
                              void* d_out, int out_size, void* d_ws, size_t ws_size,
                              hipStream_t stream) {
    const float* x     = (const float*)d_in[0];
    const float* w_qkv = (const float*)d_in[1];
    const float* w_out = (const float*)d_in[2];
    const float* b_out = (const float*)d_in[3];
    float* out = (float*)d_out;

    unsigned short* ws   = (unsigned short*)d_ws;
    unsigned short* xbf  = ws;                                   // 4096*1024
    unsigned short* wqkv = xbf  + (size_t)MTOT * DIM;            // 3072*1024
    unsigned short* wout = wqkv + (size_t)3 * DIM * DIM;         // 1024*1024
    unsigned short* qb   = wout + (size_t)DIM * DIM;             // 4096*1024
    unsigned short* kb   = qb   + (size_t)MTOT * DIM;
    unsigned short* vb   = kb   + (size_t)MTOT * DIM;
    unsigned short* ob   = vb   + (size_t)MTOT * DIM;            // 24M elems = 48 MiB

    int n;
    n = MTOT * DIM;    cvt_f32_bf16<<<(n + 255) / 256, 256, 0, stream>>>(x, xbf, n);
    n = 3 * DIM * DIM; cvt_f32_bf16<<<(n + 255) / 256, 256, 0, stream>>>(w_qkv, wqkv, n);
    n = DIM * DIM;     cvt_f32_bf16<<<(n + 255) / 256, 256, 0, stream>>>(w_out, wout, n);

    dim3 g1(MTOT / 128, (3 * DIM) / 64);
    gemm_qkv<<<g1, 256, 0, stream>>>(xbf, wqkv, qb, kb, vb);

    attn_kernel<<<(BATCH * HEADS * (SEQ / 16)) / 4, 128, 0, stream>>>(qb, kb, vb, ob);

    dim3 g2(MTOT / 128, DIM / 64);
    gemm_out<<<g2, 256, 0, stream>>>(ob, wout, b_out, out);
}
